// DepthNet_86208583565797
// MI455X (gfx1250) — matrix operand order repaired and checked by reading, not compiled, not run on hardware
//
#include <hip/hip_runtime.h>
#include <hip/hip_bf16.h>

typedef __attribute__((ext_vector_type(16))) __bf16 v16bf;
typedef __attribute__((ext_vector_type(8)))  float  v8f;

// Problem constants (match reference module-level shapes)
constexpr int Vv = 3, Cc = 32, Dd = 48, Hh = 128, Ww = 160;
constexpr int IMG = Hh * Ww;

// Tile geometry: output tile 4(d) x 4(h) x 16(w), halo +1 each side
constexpr int DT = 4, HT = 4, WT = 16;
constexpr int HD = DT + 2, HH = HT + 2, HW = WT + 2;     // 6 x 6 x 18
constexpr int NHALO = HD * HH * HW;                      // 648 voxels
constexpr int NTAP = 27;

// ---------------------------------------------------------------------------
// 4x4 inverse (adjugate)
// ---------------------------------------------------------------------------
__device__ inline void inv4x4(const float m[16], float out[16]) {
    float inv[16];
    inv[0]  =  m[5]*m[10]*m[15] - m[5]*m[11]*m[14] - m[9]*m[6]*m[15] + m[9]*m[7]*m[14] + m[13]*m[6]*m[11] - m[13]*m[7]*m[10];
    inv[4]  = -m[4]*m[10]*m[15] + m[4]*m[11]*m[14] + m[8]*m[6]*m[15] - m[8]*m[7]*m[14] - m[12]*m[6]*m[11] + m[12]*m[7]*m[10];
    inv[8]  =  m[4]*m[9]*m[15]  - m[4]*m[11]*m[13] - m[8]*m[5]*m[15] + m[8]*m[7]*m[13] + m[12]*m[5]*m[11] - m[12]*m[7]*m[9];
    inv[12] = -m[4]*m[9]*m[14]  + m[4]*m[10]*m[13] + m[8]*m[5]*m[14] - m[8]*m[6]*m[13] - m[12]*m[5]*m[10] + m[12]*m[6]*m[9];
    inv[1]  = -m[1]*m[10]*m[15] + m[1]*m[11]*m[14] + m[9]*m[2]*m[15] - m[9]*m[3]*m[14] - m[13]*m[2]*m[11] + m[13]*m[3]*m[10];
    inv[5]  =  m[0]*m[10]*m[15] - m[0]*m[11]*m[14] - m[8]*m[2]*m[15] + m[8]*m[3]*m[14] + m[12]*m[2]*m[11] - m[12]*m[3]*m[10];
    inv[9]  = -m[0]*m[9]*m[15]  + m[0]*m[11]*m[13] + m[8]*m[1]*m[15] - m[8]*m[3]*m[13] - m[12]*m[1]*m[11] + m[12]*m[3]*m[9];
    inv[13] =  m[0]*m[9]*m[14]  - m[0]*m[10]*m[13] - m[8]*m[1]*m[14] + m[8]*m[2]*m[13] + m[12]*m[1]*m[10] - m[12]*m[2]*m[9];
    inv[2]  =  m[1]*m[6]*m[15]  - m[1]*m[7]*m[14]  - m[5]*m[2]*m[15] + m[5]*m[3]*m[14] + m[13]*m[2]*m[7]  - m[13]*m[3]*m[6];
    inv[6]  = -m[0]*m[6]*m[15]  + m[0]*m[7]*m[14]  + m[4]*m[2]*m[15] - m[4]*m[3]*m[14] - m[12]*m[2]*m[7]  + m[12]*m[3]*m[6];
    inv[10] =  m[0]*m[5]*m[15]  - m[0]*m[7]*m[13]  - m[4]*m[1]*m[15] + m[4]*m[3]*m[13] + m[12]*m[1]*m[7]  - m[12]*m[3]*m[5];
    inv[14] = -m[0]*m[5]*m[14]  + m[0]*m[6]*m[13]  + m[4]*m[1]*m[14] - m[4]*m[2]*m[13] - m[12]*m[1]*m[6]  + m[12]*m[2]*m[5];
    inv[3]  = -m[1]*m[6]*m[11]  + m[1]*m[7]*m[10]  + m[5]*m[2]*m[11] - m[5]*m[3]*m[10] - m[9]*m[2]*m[7]   + m[9]*m[3]*m[6];
    inv[7]  =  m[0]*m[6]*m[11]  - m[0]*m[7]*m[10]  - m[4]*m[2]*m[11] + m[4]*m[3]*m[10] + m[8]*m[2]*m[7]   - m[8]*m[3]*m[6];
    inv[11] = -m[0]*m[5]*m[11]  + m[0]*m[7]*m[9]   + m[4]*m[1]*m[11] - m[4]*m[3]*m[9]  - m[8]*m[1]*m[7]   + m[8]*m[3]*m[5];
    inv[15] =  m[0]*m[5]*m[10]  - m[0]*m[6]*m[9]   - m[4]*m[1]*m[10] + m[4]*m[2]*m[9]  + m[8]*m[1]*m[6]   - m[8]*m[2]*m[5];
    float det = m[0]*inv[0] + m[1]*inv[4] + m[2]*inv[8] + m[3]*inv[12];
    det = 1.0f / det;
    #pragma unroll
    for (int i = 0; i < 16; ++i) out[i] = inv[i] * det;
}

// ---------------------------------------------------------------------------
// Kernel 0: per-view rot/trans of (src_proj @ inv(ref_proj)) -> ws
// ---------------------------------------------------------------------------
__global__ void mvs_proj_kernel(const float* __restrict__ proj, float* __restrict__ rt) {
    int v = threadIdx.x;
    if (v >= Vv) return;
    float refm[16], refi[16];
    #pragma unroll
    for (int i = 0; i < 16; ++i) refm[i] = proj[i];         // view 0 (b=0)
    inv4x4(refm, refi);
    const float* sp = proj + v * 16;
    float P[16];
    #pragma unroll
    for (int r = 0; r < 4; ++r)
        #pragma unroll
        for (int c = 0; c < 4; ++c) {
            float s = 0.f;
            #pragma unroll
            for (int k = 0; k < 4; ++k) s += sp[r*4 + k] * refi[k*4 + c];
            P[r*4 + c] = s;
        }
    float* o = rt + v * 12;
    o[0] = P[0];  o[1] = P[1];  o[2] = P[2];
    o[3] = P[4];  o[4] = P[5];  o[5] = P[6];
    o[6] = P[8];  o[7] = P[9];  o[8] = P[10];
    o[9] = P[3];  o[10] = P[7]; o[11] = P[11];
}

// ---------------------------------------------------------------------------
// Kernel 1: fused warp + variance (LDS, bf16) + 3x3x3 conv via WMMA -> cost
// ---------------------------------------------------------------------------
__global__ __launch_bounds__(256)
void mvs_cost_kernel(const float* __restrict__ feat,   // (V,C,H,W)
                     const float* __restrict__ rt,     // (V,12)
                     const float* __restrict__ dvals,  // (D)
                     const float* __restrict__ regw,   // (C,3,3,3)
                     const float* __restrict__ regb,   // (1)
                     float* __restrict__ cost) {       // (D,H,W)
    __shared__ __bf16 varL[NHALO * Cc]  __attribute__((aligned(64))); // 41472 B
    __shared__ __bf16 aW[NTAP * 32]     __attribute__((aligned(64))); // 1728 B, WMMA-A layout, contiguous K
    __shared__ __bf16 zeroA[NTAP * 32]  __attribute__((aligned(64))); // 1728 B of zeros

    const int tid = threadIdx.x;
    const int w0 = blockIdx.x * WT;
    const int h0 = blockIdx.y * HT;
    const int d0 = blockIdx.z * DT;

    // hint: features are L2-resident; pull tile's ref rows toward the WGP
    if (tid == 0) __builtin_prefetch(feat + h0 * Ww + w0, 0, 1);

    // Stage tap weights in WMMA-A layout + a zero region for non-row-0 lanes.
    // K order used here (A and B alike): lane half h, element j -> K = h*16 + j
    for (int i = tid; i < NTAP * 32; i += 256) {
        int tap = i >> 5, slot = i & 31;
        int half = slot >> 4, j = slot & 15;
        int ch = half * 16 + j;
        int kd = tap / 9, kh = (tap / 3) % 3, kw = tap % 3;
        aW[i] = (__bf16)regw[((ch * 3 + kd) * 3 + kh) * 3 + kw];
        zeroA[i] = (__bf16)0.0f;
    }

    // uniform per-view rot/trans (views 1..2)
    float R[2][12];
    #pragma unroll
    for (int v = 0; v < 2; ++v)
        #pragma unroll
        for (int k = 0; k < 12; ++k) R[v][k] = rt[(v + 1) * 12 + k];

    // ---------------- Phase 1: variance volume for tile + halo ----------------
    for (int i = tid; i < NHALO; i += 256) {
        int wx = i % HW, hy = (i / HW) % HH, dz = i / (HW * HH);
        int d = d0 + dz - 1, h = h0 + hy - 1, w = w0 + wx - 1;
        if (d < 0 || d >= Dd || h < 0 || h >= Hh || w < 0 || w >= Ww) {
            #pragma unroll
            for (int c = 0; c < Cc; ++c) varL[i * Cc + c] = (__bf16)0.0f;
            continue;
        }
        float dep = dvals[d];
        int   off[2][4];
        float wt[2][4];
        #pragma unroll
        for (int v = 0; v < 2; ++v) {
            float x = (float)w, y = (float)h;
            float rx = R[v][0]*x + R[v][1]*y + R[v][2];
            float ry = R[v][3]*x + R[v][4]*y + R[v][5];
            float rz = R[v][6]*x + R[v][7]*y + R[v][8];
            float px = rx * dep + R[v][9];
            float py = ry * dep + R[v][10];
            float pz = rz * dep + R[v][11];
            float gx = px / pz, gy = py / pz;
            float x0f = floorf(gx), y0f = floorf(gy);
            float fx = gx - x0f, fy = gy - y0f;
            float x1f = x0f + 1.0f, y1f = y0f + 1.0f;
            bool vx0 = (x0f >= 0.f) & (x0f <= (float)(Ww - 1));
            bool vx1 = (x1f >= 0.f) & (x1f <= (float)(Ww - 1));
            bool vy0 = (y0f >= 0.f) & (y0f <= (float)(Hh - 1));
            bool vy1 = (y1f >= 0.f) & (y1f <= (float)(Hh - 1));
            int xc0 = (int)fminf(fmaxf(x0f, 0.f), (float)(Ww - 1));
            int xc1 = (int)fminf(fmaxf(x1f, 0.f), (float)(Ww - 1));
            int yc0 = (int)fminf(fmaxf(y0f, 0.f), (float)(Hh - 1));
            int yc1 = (int)fminf(fmaxf(y1f, 0.f), (float)(Hh - 1));
            off[v][0] = yc0 * Ww + xc0;  wt[v][0] = (vx0 & vy0) ? (1.f - fx) * (1.f - fy) : 0.f;
            off[v][1] = yc0 * Ww + xc1;  wt[v][1] = (vx1 & vy0) ? fx * (1.f - fy)         : 0.f;
            off[v][2] = yc1 * Ww + xc0;  wt[v][2] = (vx0 & vy1) ? (1.f - fx) * fy         : 0.f;
            off[v][3] = yc1 * Ww + xc1;  wt[v][3] = (vx1 & vy1) ? fx * fy                 : 0.f;
        }
        const int refBase = h * Ww + w;
        constexpr float inv3 = 1.0f / 3.0f;
        #pragma unroll 4
        for (int c = 0; c < Cc; ++c) {
            float refv = feat[c * IMG + refBase];
            float s = refv, sq = refv * refv;
            #pragma unroll
            for (int v = 0; v < 2; ++v) {
                const float* fb = feat + ((v + 1) * Cc + c) * IMG;
                float wv = fb[off[v][0]] * wt[v][0] + fb[off[v][1]] * wt[v][1]
                         + fb[off[v][2]] * wt[v][2] + fb[off[v][3]] * wt[v][3];
                s += wv; sq += wv * wv;
            }
            float mean = s * inv3;
            varL[i * Cc + c] = (__bf16)(sq * inv3 - mean * mean);
        }
    }
    __syncthreads();

    // ---------------- Phase 2: 27-tap conv as WMMA accumulation ----------------
    const int lane = tid & 31;
    const int wave = tid >> 5;
    const float bias = regb[0];
    const int laneHalf = lane >> 4;              // 0 or 1
    const int laneCol  = lane & 15;

    // Branch-free A base: lanes 0/16 read real weights, all others read zeros.
    // Per-tap loads are then base + compile-time constant (DS immediate offset).
    const __bf16* aBase = (laneCol == 0) ? (aW + (laneHalf << 4)) : zeroA;

    // Two output (d,h)-rows per wave; one A load feeds two WMMAs.
    const int g0 = wave, g1 = wave + 8;
    const int dz0 = g0 >> 2, hy0 = g0 & 3;
    const int dz1 = g1 >> 2, hy1 = g1 & 3;
    const __bf16* b0 = &varL[((dz0 * HH + hy0) * HW + laneCol) * Cc + (laneHalf << 4)];
    const __bf16* b1 = &varL[((dz1 * HH + hy1) * HW + laneCol) * Cc + (laneHalf << 4)];

    v8f acc0 = {}, acc1 = {};
    #pragma unroll
    for (int kd = 0; kd < 3; ++kd)
        #pragma unroll
        for (int kh = 0; kh < 3; ++kh)
            #pragma unroll
            for (int kw = 0; kw < 3; ++kw) {
                const int tap  = (kd * 3 + kh) * 3 + kw;
                const int boff = ((kd * HH + kh) * HW + kw) * Cc;
                v16bf a   = *(const v16bf*)(aBase + tap * 32);
                v16bf vb0 = *(const v16bf*)(b0 + boff);
                v16bf vb1 = *(const v16bf*)(b1 + boff);
                acc0 = __builtin_amdgcn_wmma_f32_16x16x32_bf16(
                           false, a, false, vb0, (short)0, acc0, false, false);
                acc1 = __builtin_amdgcn_wmma_f32_16x16x32_bf16(
                           false, a, false, vb1, (short)0, acc1, false, false);
            }

    if (lane < 16) {                             // D row 0 = VGPR0, lanes 0..15
        int w = w0 + lane;
        cost[((d0 + dz0) * Hh + (h0 + hy0)) * Ww + w] = acc0[0] + bias;
        cost[((d0 + dz1) * Hh + (h0 + hy1)) * Ww + w] = acc1[0] + bias;
    }
}

// ---------------------------------------------------------------------------
// Kernel 2: softmax over depth + depth regression + confidence
// ---------------------------------------------------------------------------
__global__ __launch_bounds__(256)
void mvs_depth_kernel(const float* __restrict__ cost,
                      const float* __restrict__ dvals,
                      float* __restrict__ out) {
    int pix = blockIdx.x * 256 + threadIdx.x;
    if (pix >= IMG) return;
    float p[Dd];
    float m = -3.4e38f;
    #pragma unroll
    for (int d = 0; d < Dd; ++d) { p[d] = cost[d * IMG + pix]; m = fmaxf(m, p[d]); }
    float s = 0.f;
    #pragma unroll
    for (int d = 0; d < Dd; ++d) { p[d] = __expf(p[d] - m); s += p[d]; }
    float invs = 1.0f / s;
    float depth = 0.f, didxf = 0.f;
    #pragma unroll
    for (int d = 0; d < Dd; ++d) {
        p[d] *= invs;
        depth += p[d] * dvals[d];
        didxf += p[d] * (float)d;
    }
    int didx = (int)didxf;                  // truncation (didxf >= 0), matches astype(int32)
    didx = min(max(didx, 0), Dd - 1);
    float conf = 0.f;
    #pragma unroll
    for (int d = 0; d < Dd; ++d)
        if (d >= didx - 1 && d <= didx + 2) conf += p[d];
    out[pix] = depth;
    out[IMG + pix] = conf;
}

// ---------------------------------------------------------------------------
extern "C" void kernel_launch(void* const* d_in, const int* in_sizes, int n_in,
                              void* d_out, int out_size, void* d_ws, size_t ws_size,
                              hipStream_t stream) {
    (void)in_sizes; (void)n_in; (void)out_size; (void)ws_size;
    const float* feat  = (const float*)d_in[0];  // (1,3,32,128,160)
    const float* proj  = (const float*)d_in[1];  // (1,3,4,4)
    const float* dvals = (const float*)d_in[2];  // (1,48)
    const float* regw  = (const float*)d_in[3];  // (1,32,3,3,3)
    const float* regb  = (const float*)d_in[4];  // (1,)
    float* out = (float*)d_out;                  // depth(1,128,160) ++ conf(1,128,160)

    float* rt   = (float*)d_ws;                  // 36 floats
    float* cost = (float*)d_ws + 64;             // 256B offset; D*H*W f32 = 3.93 MB

    mvs_proj_kernel<<<1, 32, 0, stream>>>(proj, rt);

    dim3 grid(Ww / WT, Hh / HT, Dd / DT);        // (10, 32, 12) tiles
    mvs_cost_kernel<<<grid, 256, 0, stream>>>(feat, rt, dvals, regw, regb, cost);

    mvs_depth_kernel<<<(IMG + 255) / 256, 256, 0, stream>>>(cost, dvals, out);
}
